// model_LSTM_12128987643953
// MI455X (gfx1250) — compile-verified, matrix operand order checked by reading, not run on hardware
//
#include <hip/hip_runtime.h>
#include <stdint.h>

// ---------------- problem sizes ----------------
#define INP     128
#define HID     512
#define BATCH   256
#define T_STEPS 512
#define G4      2048                 // 4*HID gate columns

// ---------------- persistent kernel config ----------------
#define TPB     256                  // 8 waves of 32
#define NWG_L   32                   // workgroups per layer
#define NWG     (2*NWG_L)            // both layers pipelined
#define WG_M    128                  // batch rows per WG (2 M-tiles over B=256)
#define WG_N    128                  // gate cols per WG = 4 gates x 32 hidden units
#define KC      64                   // K-chunk staged into LDS
#define K0      (HID+INP)            // 640:  A = [h0 | x_t],  B = [W_hh0 | W_ih0]
#define K1      (HID+HID)            // 1024: A = [h2 | h1_t], B = [W_hh1 | W_ih1]
#define ABUF_H  (WG_M*KC)            // halves per A buffer (16 KB)
#define SMEM_BYTES (2*ABUF_H*2 + WG_N*K1*2)  // 32 KB A dbl-buf + 256 KB resident B = 288 KB

typedef __attribute__((ext_vector_type(16))) _Float16 v16h;
typedef __attribute__((ext_vector_type(8)))  float    v8f;
typedef int v4i __attribute__((vector_size(16)));     // matches async builtin param type
union FragH { v16h v; uint4 q[2]; };

#ifndef __has_builtin
#define __has_builtin(x) 0
#endif
#if __has_builtin(__builtin_amdgcn_global_load_async_to_lds_b128)
#define HAVE_ASYNC_LDS 1
#define AS1P(p) ((__attribute__((address_space(1))) v4i*)(uintptr_t)(p))
#define AS3P(p) ((__attribute__((address_space(3))) v4i*)(unsigned)(uintptr_t)(p))
#if __has_builtin(__builtin_amdgcn_s_wait_asynccnt)
#define ASYNC_WAIT() __builtin_amdgcn_s_wait_asynccnt(0)
#else
#define ASYNC_WAIT() asm volatile("s_wait_asynccnt 0" ::: "memory")
#endif
#else
#define HAVE_ASYNC_LDS 0
#define ASYNC_WAIT() ((void)0)
#endif

#define WMMA_F16(a, b, c) __builtin_amdgcn_wmma_f32_16x16x32_f16( \
    false, (a), false, (b), (short)0, (c), false, false)

__device__ __forceinline__ float fsig(float x)  { return 1.0f / (1.0f + __expf(-x)); }
__device__ __forceinline__ float ftanh(float x) { return 2.0f * fsig(2.0f * x) - 1.0f; }

// ---------------- helper kernels ----------------
__global__ void zero_u32(unsigned* __restrict__ p, int n) {
  for (int i = blockIdx.x * blockDim.x + threadIdx.x; i < n; i += gridDim.x * blockDim.x)
    p[i] = 0u;
}

__global__ void cvt_f32_f16(const float* __restrict__ s, _Float16* __restrict__ d, int n) {
  for (int i = blockIdx.x * blockDim.x + threadIdx.x; i < n; i += gridDim.x * blockDim.x)
    d[i] = (_Float16)s[i];
}

// Wcat[g][0:kh) = Whh[g][:], Wcat[g][kh:kh+ki) = Wih[g][:]   (f16, row-major)
__global__ void pack_w(const float* __restrict__ whh, const float* __restrict__ wih,
                       _Float16* __restrict__ dst, int kh, int ki) {
  const int K = kh + ki;
  const long n = (long)G4 * K;
  for (long i = blockIdx.x * blockDim.x + threadIdx.x; i < n; i += (long)gridDim.x * blockDim.x) {
    int g = (int)(i / K), c = (int)(i - (long)g * K);
    float v = (c < kh) ? whh[(long)g * kh + c] : wih[(long)g * ki + (c - kh)];
    dst[i] = (_Float16)v;
  }
}

__global__ void vec_add(const float* __restrict__ a, const float* __restrict__ b,
                        float* __restrict__ o, int n) {
  int i = blockIdx.x * blockDim.x + threadIdx.x;
  if (i < n) o[i] = a[i] + b[i];
}

// out[b] = dot(h[b,:], Wfc) + bfc
__global__ void fc_kernel(const _Float16* __restrict__ h, const float* __restrict__ wfc,
                          const float* __restrict__ bfc, float* __restrict__ out) {
  const int b = threadIdx.x;
  float acc = 0.f;
  for (int j = 0; j < HID; ++j) acc += (float)h[b * HID + j] * wfc[j];
  out[b] = acc + bfc[0];
}

// ---------------- persistent 2-layer LSTM ----------------
// Grid: 64 WGs x 256 thr. WGs [0,32): layer0 at t=s; WGs [32,64): layer1 at t=s-1.
// Per WG: C tile 128(batch) x 128(gate cols), gate cols = {g*512 + nwg*32 + j}.
// Weights slice LDS-resident for all 512 steps; A ([h|x]) staged per K-chunk.
__global__ __launch_bounds__(TPB, 1) void lstm_persistent(
    const _Float16* __restrict__ x16,     // [B][T][INP] f16
    const _Float16* __restrict__ Wcat0,   // [2048][640] f16
    const _Float16* __restrict__ Wcat1,   // [2048][1024] f16
    const float* __restrict__ bsum0,      // [2048]
    const float* __restrict__ bsum1,      // [2048]
    _Float16* hs0, _Float16* hs1,         // [2][B][HID] double-buffered hidden state
    _Float16* hseq,                       // [2][B][HID] layer0 -> layer1 handoff
    float* c0, float* c1,                 // [B][HID] cell state
    float* gbuf,                          // [NWG][WG_M][WG_N] f32 gate scratch
    unsigned* bar)                        // grid barrier counter (monotonic)
{
  extern __shared__ char smem[];
  _Float16* Ab = (_Float16*)smem;                      // [2][WG_M][KC]
  _Float16* Br = (_Float16*)(smem + 2 * ABUF_H * 2);   // [WG_N][Kl] resident weights

  const int tid    = threadIdx.x;
  const int lane   = tid & 31;
  const int wid    = tid >> 5;
  const int wave_m = wid & 3;            // 4 slabs of 32 batch rows
  const int wave_n = wid >> 2;           // 2 slabs of 64 gate cols
  const int layer  = blockIdx.x >> 5;
  const int wg     = blockIdx.x & 31;
  const int mwg    = wg >> 4;            // 0..1
  const int nwg    = wg & 15;            // 0..15 -> hidden units [nwg*32, nwg*32+32)
  const int m0     = mwg * WG_M;
  const int Kl     = layer ? K1 : K0;
  const int nck    = Kl / KC;            // 16 or 10
  const _Float16* Wcat = layer ? Wcat1 : Wcat0;
  const float*    bsum = layer ? bsum1 : bsum0;
  _Float16* hs  = layer ? hs1 : hs0;
  float*    cst = layer ? c1 : c0;
  float* gb = gbuf + (size_t)blockIdx.x * (WG_M * WG_N);

  // ---- load resident weight slice: LDS row lr -> Wcat row (lr/32)*512 + nwg*32 + lr%32
  {
    const int qpr = Kl / 8;              // uint4 per row
    for (int q = tid; q < WG_N * qpr; q += TPB) {
      int lr = q / qpr, qc = q - lr * qpr;
      int gr = ((lr >> 5) << 9) + (nwg << 5) + (lr & 31);
      *(uint4*)(Br + (size_t)lr * Kl + qc * 8) =
          *(const uint4*)(Wcat + (size_t)gr * Kl + qc * 8);
    }
  }
  __syncthreads();

  v8f acc[2][4];

  for (int s = 0; s <= T_STEPS; ++s) {
    const bool active = layer ? (s >= 1) : (s < T_STEPS);
    if (active) {
      const int t = layer ? (s - 1) : s;
      const _Float16* hrec = hs + (size_t)(s & 1) * (BATCH * HID);

      #pragma unroll
      for (int mi = 0; mi < 2; ++mi)
        #pragma unroll
        for (int ni = 0; ni < 4; ++ni)
          #pragma unroll
          for (int v = 0; v < 8; ++v) acc[mi][ni][v] = 0.f;

      // stage one 128x64 f16 A-chunk into LDS buffer `buf`
      auto stage = [&](int ck, int buf) {
        _Float16* dst = Ab + buf * ABUF_H;
        const _Float16* srcb; size_t rstride; int coff;
        if (ck < 8)           { srcb = hrec;                                      rstride = HID;                  coff = ck * KC; }
        else if (layer == 0)  { srcb = x16 + (size_t)t * INP;                     rstride = (size_t)T_STEPS*INP;  coff = (ck - 8) * KC; }
        else                  { srcb = hseq + (size_t)(t & 1) * (BATCH * HID);    rstride = HID;                  coff = (ck - 8) * KC; }
        for (int q = tid; q < WG_M * 8; q += TPB) {       // 8 x uint4 per row
          int r = q >> 3, qc = q & 7;
          const _Float16* src = srcb + (size_t)(m0 + r) * rstride + coff + qc * 8;
          _Float16* d = dst + r * KC + qc * 8;
#if HAVE_ASYNC_LDS
          __builtin_amdgcn_global_load_async_to_lds_b128(AS1P(src), AS3P(d), 0, 0);
#else
          *(uint4*)d = *(const uint4*)src;
#endif
        }
      };

      // ISA f16 A/B fragment: lane<16 -> K[0:8)+[16:24), lane>=16 -> K[8:16)+[24:32)
      auto ldA = [&](int rowbase, int kk, int buf) -> v16h {
        const _Float16* p = Ab + buf * ABUF_H
                          + (rowbase + (lane & 15)) * KC + kk + ((lane >> 4) << 3);
        FragH f; f.q[0] = *(const uint4*)p; f.q[1] = *(const uint4*)(p + 16);
        return f.v;
      };
      auto ldB = [&](int lrbase, int kcol) -> v16h {
        const _Float16* p = Br + (size_t)(lrbase + (lane & 15)) * Kl
                          + kcol + ((lane >> 4) << 3);
        FragH f; f.q[0] = *(const uint4*)p; f.q[1] = *(const uint4*)(p + 16);
        return f.v;
      };

      stage(0, 0);
      for (int ck = 0; ck < nck; ++ck) {
        ASYNC_WAIT();
        __syncthreads();                     // chunk ck resident in LDS
        if (ck + 1 < nck) stage(ck + 1, (ck + 1) & 1);
        const int buf = ck & 1;
        #pragma unroll
        for (int kk = 0; kk < KC; kk += 32) {
          const int kcol = ck * KC + kk;
          // preload ALL fragments for this kk so the ds_loads can overlap and
          // the scheduler can use partial s_wait_dscnt instead of wait-to-zero
          v16h a0 = ldA(wave_m * 32 +  0, kk, buf);
          v16h a1 = ldA(wave_m * 32 + 16, kk, buf);
          v16h b0 = ldB(wave_n * 64 +  0, kcol);
          v16h b1 = ldB(wave_n * 64 + 16, kcol);
          v16h b2 = ldB(wave_n * 64 + 32, kcol);
          v16h b3 = ldB(wave_n * 64 + 48, kcol);
          acc[0][0] = WMMA_F16(a0, b0, acc[0][0]);
          acc[1][0] = WMMA_F16(a1, b0, acc[1][0]);
          acc[0][1] = WMMA_F16(a0, b1, acc[0][1]);
          acc[1][1] = WMMA_F16(a1, b1, acc[1][1]);
          acc[0][2] = WMMA_F16(a0, b2, acc[0][2]);
          acc[1][2] = WMMA_F16(a1, b2, acc[1][2]);
          acc[0][3] = WMMA_F16(a0, b3, acc[0][3]);
          acc[1][3] = WMMA_F16(a1, b3, acc[1][3]);
        }
        __syncthreads();                     // protect buffer reuse (WAR)
      }

      // spill gates (f32 C layout: M = v + 8*(lane>=16), N = lane&15)
      #pragma unroll
      for (int mi = 0; mi < 2; ++mi)
        #pragma unroll
        for (int ni = 0; ni < 4; ++ni) {
          int rb  = wave_m * 32 + mi * 16 + ((lane >> 4) << 3);
          int col = wave_n * 64 + ni * 16 + (lane & 15);
          #pragma unroll
          for (int v = 0; v < 8; ++v)
            gb[(size_t)(rb + v) * WG_N + col] = acc[mi][ni][v];
        }
      __threadfence();
      __syncthreads();

      // WG-local LSTM cell update: this WG owns all 4 gates of its 32 units
      _Float16* hwr = hs + (size_t)((s + 1) & 1) * (BATCH * HID);
      const int u0 = tid * 16;
      for (int u = u0; u < u0 + 16; ++u) {
        int rowl = u >> 5, j = u & 31;
        int b = m0 + rowl;
        int jg = (nwg << 5) + j;
        float gi = gb[(size_t)rowl * WG_N + j]       + bsum[jg];
        float gf = gb[(size_t)rowl * WG_N + 32 + j]  + bsum[512  + jg];
        float gg = gb[(size_t)rowl * WG_N + 64 + j]  + bsum[1024 + jg];
        float go = gb[(size_t)rowl * WG_N + 96 + j]  + bsum[1536 + jg];
        float i_ = fsig(gi), f_ = fsig(gf), g_ = ftanh(gg), o_ = fsig(go);
        size_t ci = (size_t)b * HID + jg;
        float c_ = f_ * cst[ci] + i_ * g_;
        cst[ci] = c_;
        float h_ = o_ * ftanh(c_);
        hwr[ci] = (_Float16)h_;
        if (layer == 0) hseq[(size_t)(s & 1) * (BATCH * HID) + ci] = (_Float16)h_;
      }
    }

    // ---- device-wide barrier (one per timestep), monotonic counter ----
    __threadfence();
    __syncthreads();
    if (tid == 0) {
      __hip_atomic_fetch_add(bar, 1u, __ATOMIC_ACQ_REL, __HIP_MEMORY_SCOPE_AGENT);
      const unsigned target = (unsigned)(s + 1) * (unsigned)NWG;
      while (__hip_atomic_load(bar, __ATOMIC_ACQUIRE, __HIP_MEMORY_SCOPE_AGENT) < target)
        __builtin_amdgcn_s_sleep(2);
    }
    __syncthreads();
  }
}

// ---------------- host side ----------------
extern "C" void kernel_launch(void* const* d_in, const int* in_sizes, int n_in,
                              void* d_out, int out_size, void* d_ws, size_t ws_size,
                              hipStream_t stream) {
  const float* x     = (const float*)d_in[0];
  const float* W_ih0 = (const float*)d_in[1];
  const float* W_hh0 = (const float*)d_in[2];
  const float* b_ih0 = (const float*)d_in[3];
  const float* b_hh0 = (const float*)d_in[4];
  const float* W_ih1 = (const float*)d_in[5];
  const float* W_hh1 = (const float*)d_in[6];
  const float* b_ih1 = (const float*)d_in[7];
  const float* b_hh1 = (const float*)d_in[8];
  const float* W_fc  = (const float*)d_in[9];
  const float* b_fc  = (const float*)d_in[10];
  (void)in_sizes; (void)n_in; (void)out_size; (void)ws_size;

  char* ws = (char*)d_ws;
  const size_t o_bar  = 0;                                   // 256 B (counter)
  const size_t o_hs0  = 256;                                 // 2*B*H f16 = 512 KB
  const size_t o_hs1  = o_hs0  + (size_t)2*BATCH*HID*2;
  const size_t o_c0   = o_hs1  + (size_t)2*BATCH*HID*2;      // B*H f32 = 512 KB
  const size_t o_c1   = o_c0   + (size_t)BATCH*HID*4;
  const size_t o_hseq = o_c1   + (size_t)BATCH*HID*4;        // 2*B*H f16
  const size_t o_bs0  = o_hseq + (size_t)2*BATCH*HID*2;
  const size_t o_bs1  = o_bs0  + (size_t)G4*4;
  const size_t o_w0   = o_bs1  + (size_t)G4*4;               // 2048*640  f16
  const size_t o_w1   = o_w0   + (size_t)G4*K0*2;            // 2048*1024 f16
  const size_t o_x16  = o_w1   + (size_t)G4*K1*2;            // B*T*INP   f16 (32 MB)
  const size_t o_gbuf = o_x16  + (size_t)BATCH*T_STEPS*INP*2;// 64*128*128 f32 (4 MB)

  unsigned*  bar   = (unsigned*)(ws + o_bar);
  _Float16*  hs0   = (_Float16*)(ws + o_hs0);
  _Float16*  hs1   = (_Float16*)(ws + o_hs1);
  float*     c0    = (float*)   (ws + o_c0);
  float*     c1    = (float*)   (ws + o_c1);
  _Float16*  hseq  = (_Float16*)(ws + o_hseq);
  float*     bs0   = (float*)   (ws + o_bs0);
  float*     bs1   = (float*)   (ws + o_bs1);
  _Float16*  wcat0 = (_Float16*)(ws + o_w0);
  _Float16*  wcat1 = (_Float16*)(ws + o_w1);
  _Float16*  x16   = (_Float16*)(ws + o_x16);
  float*     gbuf  = (float*)   (ws + o_gbuf);

  (void)hipFuncSetAttribute(reinterpret_cast<const void*>(lstm_persistent),
                            hipFuncAttributeMaxDynamicSharedMemorySize, SMEM_BYTES);

  // zero [bar | hs0 | hs1 | c0 | c1] (contiguous): 2,097,408 bytes
  zero_u32<<<64, 256, 0, stream>>>(bar, (int)((256 + 4*(size_t)BATCH*HID*4) / 4));
  cvt_f32_f16<<<4096, 256, 0, stream>>>(x, x16, BATCH * T_STEPS * INP);
  pack_w<<<1024, 256, 0, stream>>>(W_hh0, W_ih0, wcat0, HID, INP);
  pack_w<<<2048, 256, 0, stream>>>(W_hh1, W_ih1, wcat1, HID, HID);
  vec_add<<<8, 256, 0, stream>>>(b_ih0, b_hh0, bs0, G4);
  vec_add<<<8, 256, 0, stream>>>(b_ih1, b_hh1, bs1, G4);

  lstm_persistent<<<NWG, TPB, SMEM_BYTES, stream>>>(
      x16, wcat0, wcat1, bs0, bs1, hs0, hs1, hseq, c0, c1, gbuf, bar);

  // final h2[T-1] lives in hs1 buffer index (T+1)&1 == 1
  fc_kernel<<<1, 256, 0, stream>>>(hs1 + (size_t)BATCH * HID, W_fc, b_fc, (float*)d_out);
}
